// ConvSparseKernel_16673063043688
// MI455X (gfx1250) — matrix operand order repaired and checked by reading, not compiled, not run on hardware
//
#include <hip/hip_runtime.h>

typedef __attribute__((ext_vector_type(16))) __bf16 v16bf;
typedef __attribute__((ext_vector_type(8)))  float  v8f;

#define NTAPS  5
#define CIN    256
#define COUT   256
#define XH     64
#define XW     64
#define OH     62
#define OW     62
#define NB     16
#define KSTEPS 8                 // 256 / 32 channels per K block
#define PXS    66                // staged pixels per row: 64 + 2 (kw up to 2)
#define CST    40                // padded channel stride in bf16 elems (80B, 16B aligned)
#define LROW   (PXS * CST)       // 2640
#define LDSN   (3 * LROW)        // 7920 ushorts per plane

#define WPK_ELEMS ((size_t)NTAPS * 16 * KSTEPS * 32 * 16)   // 327680 per plane
#define XT_ELEMS  ((size_t)NB * XH * XW * CIN)              // 16777216 per plane
#define XT_BYTES  (XT_ELEMS * 2)

// ---- bf16 helpers (round-to-nearest-even split) ----
__device__ __forceinline__ unsigned short f2bf(float f) {
    unsigned u = __float_as_uint(f);
    unsigned r = (u + 0x7FFFu + ((u >> 16) & 1u)) >> 16;
    return (unsigned short)r;
}
__device__ __forceinline__ float bf2f(unsigned short h) {
    return __uint_as_float(((unsigned)h) << 16);
}

__device__ __forceinline__ v16bf ld16bf(const unsigned short* p) {
    union { uint4 q[2]; v16bf v; } u;
    u.q[0] = ((const uint4*)p)[0];
    u.q[1] = ((const uint4*)p)[1];
    return u.v;
}

__device__ __forceinline__ v8f wmma_bf16(v16bf a, v16bf b, v8f c) {
    return __builtin_amdgcn_wmma_f32_16x16x32_bf16(false, a, false, b, (short)0, c,
                                                   false, false);
}

// Async 16B global -> LDS copy (per-lane addresses), tracked by ASYNCcnt.
__device__ __forceinline__ void async_cp16(unsigned lds_byte_addr,
                                           unsigned long long gaddr) {
    asm volatile("global_load_async_to_lds_b128 %0, %1, off"
                 :
                 : "v"(lds_byte_addr), "v"(gaddr)
                 : "memory");
}
__device__ __forceinline__ void wait_async0() {
    asm volatile("s_wait_asynccnt 0x0" ::: "memory");
}

// ============================================================================
// Kernel 1: weight standardization + bf16 hi/lo split + pack into A-fragment
// layout Wpk[tap][mtile][kstep][lane][16] (one contiguous 32B chunk per lane).
// ============================================================================
__global__ void ws_prep(const float* __restrict__ W,
                        unsigned short* __restrict__ Wh,
                        unsigned short* __restrict__ Wl) {
    int wv = threadIdx.x >> 5, lane = threadIdx.x & 31;
    int row = blockIdx.x * 8 + wv;               // 0..1279 = tap*256 + out_ch
    const float* wr = W + (size_t)row * CIN;

    float v[8];
    float s = 0.f;
#pragma unroll
    for (int e = 0; e < 8; ++e) { v[e] = wr[e * 32 + lane]; s += v[e]; }
#pragma unroll
    for (int o = 16; o > 0; o >>= 1) s += __shfl_xor(s, o, 32);
    float mean = s * (1.0f / 256.0f);
    float ss = 0.f;
#pragma unroll
    for (int e = 0; e < 8; ++e) { v[e] -= mean; ss += v[e] * v[e]; }
#pragma unroll
    for (int o = 16; o > 0; o >>= 1) ss += __shfl_xor(ss, o, 32);
    float inv = 1.0f / sqrtf(ss);

    int k = row >> 8, oc = row & 255;
    int mt = oc >> 4, r = oc & 15;
#pragma unroll
    for (int e = 0; e < 8; ++e) {
        float f = v[e] * inv;
        int i = e * 32 + lane;
        int ks  = i >> 5;
        int off = i & 31;
        int lane2 = r + ((off >> 4) << 4);
        int j = off & 15;
        size_t idx = ((((size_t)(k * 16 + mt) * KSTEPS + ks) * 32 + lane2) << 4) + j;
        unsigned short hu = f2bf(f);
        Wh[idx] = hu;
        Wl[idx] = f2bf(f - bf2f(hu));
    }
}

// ============================================================================
// Kernel 2: one-time x transpose+convert: [b][c][h][w] f32 ->
// channels-last bf16 hi/lo planes [b][h][w][c].
// ============================================================================
__global__ void xt_prep(const float* __restrict__ x,
                        unsigned short* __restrict__ xTH,
                        unsigned short* __restrict__ xTL) {
    __shared__ float tile[32][33];
    int bid = blockIdx.x;                 // 16384 blocks
    int ct = bid & 7;
    int wt = (bid >> 3) & 1;
    int bh = bid >> 4;                    // 0..1023 = b*64 + h
    int b = bh >> 6, h = bh & 63;

    int w  = threadIdx.x & 31;
    int cq = threadIdx.x >> 5;
#pragma unroll
    for (int i = 0; i < 4; ++i) {
        int cl = cq + i * 8;
        float f = x[(((size_t)(b * 256 + ct * 32 + cl) * 64 + h) << 6) + wt * 32 + w];
        tile[cl][w] = f;
    }
    __syncthreads();
    int cl = threadIdx.x & 31;
    int wq = threadIdx.x >> 5;
#pragma unroll
    for (int j = 0; j < 4; ++j) {
        int wl = wq + j * 8;
        float f = tile[cl][wl];
        unsigned short hu = f2bf(f);
        size_t o = ((((size_t)(b * 64 + h) << 6) + wt * 32 + wl) << 8) + ct * 32 + cl;
        xTH[o] = hu;
        xTL[o] = f2bf(f - bf2f(hu));
    }
}

// ============================================================================
// Kernel 3: main conv-as-GEMM. Block = (batch, output row). M=256 out-ch,
// N=64 pixels, K = 5 taps x 256 in-ch folded. BF16x3 FP32 emulation.
// ASYNC=true : double-buffered async DMA staging with hoisted per-thread copy
//              slots; copies for ks+1 overlap the 120 WMMAs of ks.
// ASYNC=false: fallback, stage from f32 x with VALU conversion.
// ============================================================================
template <bool ASYNC>
__global__ __launch_bounds__(256) void conv_wmma(
        const float* __restrict__ x,
        const unsigned short* __restrict__ xTH,
        const unsigned short* __restrict__ xTL,
        const unsigned short* __restrict__ Wh,
        const unsigned short* __restrict__ Wl,
        const float* __restrict__ bias,
        float* __restrict__ out) {
    constexpr int NBUF = ASYNC ? 2 : 1;
    __shared__ __align__(16) unsigned short smem[NBUF * 2 * LDSN];

    const int KH[NTAPS] = {0, 0, 1, 2, 2};
    const int KW[NTAPS] = {0, 2, 1, 0, 2};

    int tid = threadIdx.x, lane = tid & 31, wv = tid >> 5;
    int hb = blockIdx.x % OH;
    int b  = blockIdx.x / OH;

    const v8f vzero = {0.f, 0.f, 0.f, 0.f, 0.f, 0.f, 0.f, 0.f};
    v8f acc[2][4];
#pragma unroll
    for (int m = 0; m < 2; ++m)
#pragma unroll
        for (int n = 0; n < 4; ++n) acc[m][n] = vzero;

    const float* xb = x + (size_t)b * CIN * (XH * XW) + (size_t)hb * XW;

    // --- per-thread async copy slots, invariant across K-steps (<= 4) ---
    int nslot = 0;
    unsigned la_off[4];             // LDS byte offset within a buffer (hi plane)
    unsigned long long gh[4];       // global byte address in xTH at ks = 0
    if constexpr (ASYNC) {
        for (int e = tid; e < 3 * PXS * 4; e += 256) {
            int chunk = e & 3;
            int t  = e >> 2;
            int px = t % PXS;
            int row = t / PXS;
            int wx = px > (XW - 1) ? (XW - 1) : px;
            size_t goff = ((((size_t)(b * XH + hb + row) << 6) + wx) << 8)
                          + chunk * 8;                        // ushort elems
            la_off[nslot] = (unsigned)((row * LROW + px * CST + chunk * 8) * 2);
            gh[nslot] = (unsigned long long)(uintptr_t)(xTH + goff);
            ++nslot;
        }
    }
    unsigned smem_base = (unsigned)(uintptr_t)&smem[0];

    // Issue the async slab copies for channel block `ks` into buffer `bufoff`.
    auto stage_async = [&](int ks, int bufoff_elems) {
        unsigned bo = smem_base + (unsigned)(bufoff_elems * 2);
        unsigned long long kadd = (unsigned long long)(ks * 64); // 32 ch * 2B
        for (int i = 0; i < nslot; ++i) {
            unsigned la = bo + la_off[i];
            unsigned long long g = gh[i] + kadd;
            async_cp16(la, g);                                   // hi plane
            async_cp16(la + (unsigned)(LDSN * 2), g + XT_BYTES); // lo plane
        }
    };

    // 5 taps x (2 M-tiles x 4 N-tiles) x BF16x3 = 120 WMMAs per K-step.
    auto compute_step = [&](int ks, const unsigned short* sH,
                            const unsigned short* sL) {
#pragma unroll
        for (int k = 0; k < NTAPS; ++k) {
            size_t ab = ((((size_t)k * 16 + 2 * wv) * KSTEPS + ks) * 32 + lane) << 4;
            v16bf ah0 = ld16bf(Wh + ab);
            v16bf al0 = ld16bf(Wl + ab);
            v16bf ah1 = ld16bf(Wh + ab + (size_t)(KSTEPS * 32 * 16));
            v16bf al1 = ld16bf(Wl + ab + (size_t)(KSTEPS * 32 * 16));

            int rbase = KH[k] * LROW + ((lane >> 4) << 4);
            int pxb   = (lane & 15) + KW[k];
#pragma unroll
            for (int nt = 0; nt < 4; ++nt) {
                int la = rbase + (pxb + nt * 16) * CST;
                v16bf bh = ld16bf(sH + la);
                v16bf bl = ld16bf(sL + la);
                acc[0][nt] = wmma_bf16(ah0, bh, acc[0][nt]);
                acc[0][nt] = wmma_bf16(ah0, bl, acc[0][nt]);
                acc[0][nt] = wmma_bf16(al0, bh, acc[0][nt]);
                acc[1][nt] = wmma_bf16(ah1, bh, acc[1][nt]);
                acc[1][nt] = wmma_bf16(ah1, bl, acc[1][nt]);
                acc[1][nt] = wmma_bf16(al1, bh, acc[1][nt]);
            }
        }
    };

    if constexpr (ASYNC) {
        stage_async(0, 0);
        wait_async0();
        __syncthreads();
        for (int ks = 0; ks < KSTEPS; ++ks) {
            int cur = (ks & 1) * (2 * LDSN);
            int nxt = ((ks + 1) & 1) * (2 * LDSN);
            if (ks + 1 < KSTEPS) stage_async(ks + 1, nxt);   // DMA overlaps WMMAs
            compute_step(ks, smem + cur, smem + cur + LDSN);
            wait_async0();       // own copies into `nxt` have landed
            __syncthreads();     // everyone done reading `cur` / writing `nxt`
        }
    } else {
        for (int ks = 0; ks < KSTEPS; ++ks) {
            __syncthreads();
            for (int e = tid; e < 3 * 32 * PXS; e += 256) {
                int px  = e % PXS;
                int t   = e / PXS;
                int c   = t & 31;
                int row = t >> 5;
                int wx  = px > (XW - 1) ? (XW - 1) : px;
                float f = xb[(size_t)(ks * 32 + c) * (XH * XW) + row * XW + wx];
                unsigned short hu = f2bf(f);
                int la = row * LROW + px * CST + c;
                smem[la] = hu;
                smem[LDSN + la] = f2bf(f - bf2f(hu));
            }
            __syncthreads();
            compute_step(ks, smem, smem + LDSN);
        }
    }

    const float NORMF = 1.0f / sqrtf((float)(CIN * NTAPS + 1));
    size_t obase = (size_t)b * COUT * (OH * OW) + (size_t)hb * OW;
#pragma unroll
    for (int m = 0; m < 2; ++m) {
        int mt = 2 * wv + m;
#pragma unroll
        for (int nt = 0; nt < 4; ++nt) {
            int wpix = nt * 16 + (lane & 15);
            if (wpix < OW) {
#pragma unroll
                for (int v = 0; v < 8; ++v) {
                    int o = mt * 16 + v + ((lane >> 4) << 3);
                    float r = (acc[m][nt][v] + bias[o]) * NORMF;
                    out[obase + (size_t)o * (OH * OW) + wpix] = r;
                }
            }
        }
    }
}

extern "C" void kernel_launch(void* const* d_in, const int* in_sizes, int n_in,
                              void* d_out, int out_size, void* d_ws, size_t ws_size,
                              hipStream_t stream) {
    const float* x    = (const float*)d_in[0];
    const float* W    = (const float*)d_in[1];
    const float* bias = (const float*)d_in[2];
    float* out = (float*)d_out;

    unsigned short* Wh  = (unsigned short*)d_ws;
    unsigned short* Wl  = Wh + WPK_ELEMS;
    unsigned short* xTH = Wl + WPK_ELEMS;
    unsigned short* xTL = xTH + XT_ELEMS;

    const size_t need = (2 * WPK_ELEMS + 2 * XT_ELEMS) * sizeof(unsigned short);

    ws_prep<<<(NTAPS * COUT) / 8, 256, 0, stream>>>(W, Wh, Wl);

    if (ws_size >= need) {
        xt_prep<<<NB * XH * 2 * 8, 256, 0, stream>>>(x, xTH, xTL);
        conv_wmma<true><<<NB * OH, 256, 0, stream>>>(x, xTH, xTL, Wh, Wl, bias, out);
    } else {
        conv_wmma<false><<<NB * OH, 256, 0, stream>>>(x, xTH, xTL, Wh, Wl, bias, out);
    }
}
